// NeighborhoodAttention_28613072126650
// MI455X (gfx1250) — matrix operand order repaired and checked by reading, not compiled, not run on hardware
//
#include <hip/hip_runtime.h>

typedef __attribute__((ext_vector_type(16))) _Float16 v16h;
typedef __attribute__((ext_vector_type(8)))  _Float16 v8h;
typedef __attribute__((ext_vector_type(8)))  float    v8f;
typedef __attribute__((ext_vector_type(4)))  float    v4f;
typedef __attribute__((ext_vector_type(4)))  unsigned int v4u;

// Problem constants (match reference setup_inputs)
#define TT_   16
#define HH_   32
#define WW_   32
#define NH_   8
#define DH_   128
#define KT_   3
#define KH_   7
#define KW_   7
#define QT_   4
#define QH_   8
#define QW_   8
#define ST_   (QT_+KT_-1)            // 6
#define SH_   (QH_+KH_-1)            // 14
#define SW_   (QW_+KW_-1)            // 14
#define SVOL_ (ST_*SH_*SW_)          // 1176
#define NTT_  (TT_/QT_)
#define NTH_  (HH_/QH_)
#define NTW_  (WW_/QW_)
#define NTILES_ (NTT_*NTH_*NTW_)     // 64
#define CHUNK_ 32
#define NCHUNK_ ((SVOL_+CHUNK_-1)/CHUNK_)  // 37
#define LOG2E_  1.4426950408889634f
#define QSCALE_ (0.08838834764831845f * LOG2E_)   // 1/sqrt(128) * log2(e)
#define NEGBIG_ (-1e30f)

union HU8 { v8h h; v4u u; };

static __device__ __forceinline__ v16h mk16(v8h lo, v8h hi) {
  v16h r;
#pragma unroll
  for (int e = 0; e < 8; ++e) { r[e] = lo[e]; r[8 + e] = hi[e]; }
  return r;
}

static __device__ __forceinline__ v4u shflswap16(v4u x) {
  v4u r;
#pragma unroll
  for (int e = 0; e < 4; ++e) r[e] = (unsigned)__shfl_xor((int)x[e], 16, 32);
  return r;
}

static __device__ __forceinline__ int clampwin(int i, int X, int k) {
  int s = i - (k - 1) / 2;
  if (s < 0) s = 0;
  if (s > X - k) s = X - k;
  return s;
}

// advance halo-local (t,h,w) by 32 positions in the (6,14,14) mixed radix
static __device__ __forceinline__ void adv32(int& t, int& h, int& w) {
  w += 4; h += 2;                 // 32 = 2*14 + 4
  if (w >= 14) { w -= 14; h += 1; }
  if (h >= 14) { h -= 14; t += 1; }
}

__global__ __launch_bounds__(512)
void na3d_fwd_kernel(const float* __restrict__ qg, const float* __restrict__ kg,
                     const float* __restrict__ vg, float* __restrict__ out) {
  __shared__ _Float16 sK[CHUNK_][DH_];   // [key][d]   8 KB
  __shared__ _Float16 sVt[DH_][CHUNK_];  // [d][key]   8 KB

  const int tile = blockIdx.x;
  const int head = blockIdx.y;
  const int tid  = threadIdx.x;
  const int wave = tid >> 5;     // 0..15 -> owns queries wave*16..+15
  const int lane = tid & 31;
  const int ln16 = lane & 15;
  const int hig  = lane >> 4;

  // ---- tile decode & halo base (matches _axis_window) ----
  const int tt = tile / (NTH_ * NTW_);
  const int th = (tile / NTW_) % NTH_;
  const int tw = tile % NTW_;
  const int q0t = tt * QT_, q0h = th * QH_, q0w = tw * QW_;
  int baseT = clampwin(q0t, TT_, KT_); if (baseT > TT_ - ST_) baseT = TT_ - ST_;
  int baseH = clampwin(q0h, HH_, KH_); if (baseH > HH_ - SH_) baseH = HH_ - SH_;
  int baseW = clampwin(q0w, WW_, KW_); if (baseW > WW_ - SW_) baseW = WW_ - SW_;
  const int gbase = (baseT * HH_ + baseH) * WW_ + baseW;

  // ---- this lane's query (column of S^T) ----
  const int qloc = wave * 16 + ln16;
  const int lqt = q0t + (qloc >> 6);
  const int lqh = q0h + ((qloc >> 3) & 7);
  const int lqw = q0w + (qloc & 7);
  const int gq  = (lqt * HH_ + lqh) * WW_ + lqw;
  // per-lane halo-local window bitmasks
  const unsigned tbitsv = 0x7u  << (clampwin(lqt, TT_, KT_) - baseT);
  const unsigned hbitsv = 0x7Fu << (clampwin(lqh, HH_, KH_) - baseH);
  const unsigned wbitsv = 0x7Fu << (clampwin(lqw, WW_, KW_) - baseW);

  // ---- resident Q B-fragments (rows = d, cols = q); scale+log2e folded ----
  v16h qf[4];
  {
    const float* qrow = qg + ((size_t)gq * NH_ + head) * DH_;
#pragma unroll
    for (int kk = 0; kk < 4; ++kk) {
      const int d0 = kk * 32 + (hig << 3);
#pragma unroll
      for (int j = 0; j < 16; ++j)
        qf[kk][j] = (_Float16)(qrow[d0 + j + ((j >= 8) ? 8 : 0)] * QSCALE_);
    }
  }

  // ---- accumulators ----
  v8f o[8];
#pragma unroll
  for (int dt = 0; dt < 8; ++dt)
#pragma unroll
    for (int e = 0; e < 8; ++e) o[dt][e] = 0.f;
  float mrun = NEGBIG_, lrun = 0.f;

  // ---- staging roles & incremental halo coordinates ----
  const int stK_key = tid >> 4;          // 0..31
  const int stK_d0  = (tid & 15) * 8;    // 0..120
  const int stV_d   = tid & 127;
  const int stV_kb  = tid >> 7;          // 0..3
  int kjt = 0, kjh = stK_key / SW_, kjw = stK_key % SW_;      // j = stK_key (<32)
  int vjt = 0, vjh = (stV_kb * 8) / SW_, vjw = (stV_kb * 8) % SW_;
  // uniform chunk-base halo coords (SALU)
  int mbt = 0, mbh = 0, mbw = 0;

  const size_t headoff = (size_t)head * DH_;
  auto keyaddr = [&](int t, int h, int w) -> size_t {
    if (t > ST_ - 1) t = ST_ - 1;        // clamp padded tail (masked anyway)
    const int gk = gbase + (t << 10) + (h << 5) + w;   // (t*H + h)*W + w
    return ((size_t)gk << 10) + headoff;               // gk*NH*DH
  };

  v4f kr0, kr1;
  float vr[8];
  auto stage_load = [&]() {
    const float* kp = kg + keyaddr(kjt, kjh, kjw) + stK_d0;
    kr0 = *(const v4f*)kp;
    kr1 = *(const v4f*)(kp + 4);
    adv32(kjt, kjh, kjw);
#pragma unroll
    for (int e = 0; e < 8; ++e) {
      const int w0 = vjw + e;
      const int cw = (w0 >= 14) ? 1 : 0;
      const int h0 = vjh + cw;
      const int ch = (h0 >= 14) ? 1 : 0;
      const int tq = vjt + ch;
      const int hq = ch ? h0 - 14 : h0;
      const int wq = cw ? w0 - 14 : w0;
      vr[e] = vg[keyaddr(tq, hq, wq) + stV_d];
    }
    adv32(vjt, vjh, vjw);
  };

  stage_load();   // prologue: loads for chunk 0 in flight

  for (int c = 0; c < NCHUNK_; ++c) {
    // convert staged registers (waits on in-flight loads here)
    v8h hk, hv;
#pragma unroll
    for (int e = 0; e < 4; ++e) { hk[e] = (_Float16)kr0[e]; hk[4 + e] = (_Float16)kr1[e]; }
#pragma unroll
    for (int e = 0; e < 8; ++e) hv[e] = (_Float16)vr[e];

    __syncthreads();                           // chunk c-1 LDS reads done
    *(v8h*)(&sK[stK_key][stK_d0]) = hk;
    *(v8h*)(&sVt[stV_d][stV_kb * 8]) = hv;
    __syncthreads();                           // chunk c LDS published

    if (c + 1 < NCHUNK_) stage_load();         // overlap next chunk's HBM loads

    // ---- S^T = K · Q^T : two 16x16 tiles (keys 0..15 / 16..31) ----
    v8f s0, s1;
#pragma unroll
    for (int e = 0; e < 8; ++e) { s0[e] = 0.f; s1[e] = 0.f; }
#pragma unroll
    for (int kk = 0; kk < 4; ++kk) {
      const int dlo = kk * 32 + hig * 8;
      const _Float16* p0 = &sK[ln16][dlo];
      const _Float16* p1 = &sK[16 + ln16][dlo];
      v16h a0 = mk16(*(const v8h*)p0, *(const v8h*)(p0 + 16));
      v16h a1 = mk16(*(const v8h*)p1, *(const v8h*)(p1 + 16));
      s0 = __builtin_amdgcn_wmma_f32_16x16x32_f16(false, a0, false, qf[kk],
                                                  (short)0, s0, false, false);
      s1 = __builtin_amdgcn_wmma_f32_16x16x32_f16(false, a1, false, qf[kk],
                                                  (short)0, s1, false, false);
    }

    // ---- 32-key validity bitmask: <=4 w-rows, scalar row bookkeeping ----
    unsigned macc;
    {
      const int b0 = 14 - mbw;                 // keys in row 0 (uniform)
      macc = (wbitsv >> mbw) & (0u - ((hbitsv >> mbh) & (tbitsv >> mbt) & 1u));
      int h1 = mbh + 1, t1 = mbt; if (h1 >= 14) { h1 -= 14; t1 += 1; }
      macc |= (wbitsv << b0) & (0u - ((hbitsv >> h1) & (tbitsv >> t1) & 1u));
      int h2 = mbh + 2, t2 = mbt; if (h2 >= 14) { h2 -= 14; t2 += 1; }
      macc |= (wbitsv << (b0 + 14)) & (0u - ((hbitsv >> h2) & (tbitsv >> t2) & 1u));
      const int sh3 = b0 + 28;
      if (sh3 < 32) {                          // uniform branch
        int h3 = mbh + 3, t3 = mbt; if (h3 >= 14) { h3 -= 14; t3 += 1; }
        macc |= (wbitsv << sh3) & (0u - ((hbitsv >> h3) & (tbitsv >> t3) & 1u));
      }
    }
    const unsigned mlo = macc >> (hig << 3);   // this lane group's bits

    // ---- mask + online softmax (column = in-lane 16 + partner 16) ----
    bool okA[8], okB[8];
    float cmax = NEGBIG_;
#pragma unroll
    for (int r = 0; r < 8; ++r) {
      okA[r] = (mlo >> r) & 1u;
      okB[r] = (mlo >> (r + 16)) & 1u;
      s0[r] = okA[r] ? s0[r] : NEGBIG_;
      s1[r] = okB[r] ? s1[r] : NEGBIG_;
      cmax = fmaxf(cmax, s0[r]);
      cmax = fmaxf(cmax, s1[r]);
    }
    cmax = fmaxf(cmax, __shfl_xor(cmax, 16, 32));
    const float mnew = fmaxf(mrun, cmax);

    v8h p0h, p1h;
    float csum = 0.f;
#pragma unroll
    for (int r = 0; r < 8; ++r) {
      const float pa = okA[r] ? exp2f(s0[r] - mnew) : 0.f;
      const float pb = okB[r] ? exp2f(s1[r] - mnew) : 0.f;
      csum += pa + pb;
      p0h[r] = (_Float16)pa;
      p1h[r] = (_Float16)pb;
    }
    csum += __shfl_xor(csum, 16, 32);

    if (__ballot(mnew > mrun)) {               // uniform: skip rescale when max unchanged
      const float alpha = exp2f(mrun - mnew);
      lrun *= alpha;
#pragma unroll
      for (int dt = 0; dt < 8; ++dt)
#pragma unroll
        for (int e = 0; e < 8; ++e) o[dt][e] *= alpha;
    }
    mrun = mnew;
    lrun += csum;

    // ---- build P^T B-fragment: swap packed halves with partner lane ----
    HU8 ua, ub; ua.h = p0h; ub.h = p1h;
    const v4u xa = shflswap16(ua.u);
    const v4u xb = shflswap16(ub.u);
    HU8 flo, fhi;
    flo.u = (hig == 0) ? ua.u : xb;
    fhi.u = (hig == 0) ? xa   : ub.u;
    const v16h pb16 = mk16(flo.h, fhi.h);

    // ---- O^T += V^T · P^T (8 d-tiles) ----
#pragma unroll
    for (int dt = 0; dt < 8; ++dt) {
      const _Float16* pv = &sVt[dt * 16 + ln16][hig * 16];
      v16h av = mk16(*(const v8h*)pv, *(const v8h*)(pv + 8));
      o[dt] = __builtin_amdgcn_wmma_f32_16x16x32_f16(false, av, false, pb16,
                                                     (short)0, o[dt], false, false);
    }

    adv32(mbt, mbh, mbw);                      // uniform chunk-base update (SALU)
  }

  // ---- epilogue: normalize and store ----
  const float inv = 1.0f / lrun;
  float* orow = out + ((size_t)gq * NH_ + head) * DH_;
#pragma unroll
  for (int dt = 0; dt < 8; ++dt) {
    const int d0 = dt * 16 + hig * 8;
    v4f w0, w1;
#pragma unroll
    for (int e = 0; e < 4; ++e) {
      w0[e] = o[dt][e] * inv;
      w1[e] = o[dt][4 + e] * inv;
    }
    *(v4f*)(orow + d0)     = w0;
    *(v4f*)(orow + d0 + 4) = w1;
  }
}

extern "C" void kernel_launch(void* const* d_in, const int* in_sizes, int n_in,
                              void* d_out, int out_size, void* d_ws, size_t ws_size,
                              hipStream_t stream) {
  (void)in_sizes; (void)n_in; (void)d_ws; (void)ws_size; (void)out_size;
  const float* q = (const float*)d_in[0];
  const float* k = (const float*)d_in[1];
  const float* v = (const float*)d_in[2];
  float* out = (float*)d_out;
  dim3 grid(NTILES_, NH_, 1);
  dim3 block(512, 1, 1);
  na3d_fwd_kernel<<<grid, block, 0, stream>>>(q, k, v, out);
}
